// CausalSelfAttention_87041807220864
// MI455X (gfx1250) — matrix operand order repaired and checked by reading, not compiled, not run on hardware
//
#include <hip/hip_runtime.h>

#define B_ 4
#define T_ 2048
#define H_ 1024
#define NH_ 16
#define DH_ 64
#define R_ 8
#define LSCALE 2.0f

typedef __bf16 bf16;
typedef __bf16 v16bf __attribute__((ext_vector_type(16)));
typedef float  v8f   __attribute__((ext_vector_type(8)));
typedef unsigned v4u __attribute__((ext_vector_type(4)));
typedef unsigned v8u __attribute__((ext_vector_type(8)));

// 16-bit A-matrix 16x32 fragment K index for VGPR pair v, lane-half h.
__device__ __forceinline__ int kpatA(int v, int h) {
  return 2 * v + (v >= 4 ? 8 : 0) + 8 * h;
}
// 16-bit B-matrix 32x16 fragment K index, same order as A (lanes 0-15: K 0-7,16-23; lanes 16-31: K 8-15,24-31).
__device__ __forceinline__ int kpatB(int v, int h) {
  return 2 * v + (v >= 4 ? 8 : 0) + 8 * h;
}

// CDNA5 async global->LDS copy (ASYNCcnt-tracked, no VGPR data path).
__device__ __forceinline__ void async_copy_b128(void* lds_dst, const void* gsrc) {
  unsigned lds_off = (unsigned)(unsigned long long)lds_dst;
  asm volatile("global_load_async_to_lds_b128 %0, %1, off"
               :
               : "v"(lds_off), "v"(gsrc)
               : "memory");
}
__device__ __forceinline__ void wait_async0() {
  asm volatile("s_wait_asynccnt 0x0" ::: "memory");
}

// CDNA5 Tensor Data Mover: 2D tile load, bf16 elements (data_size=2B).
// D# group0: [1:0]=count=1, [63:32]=lds_addr, [120:64]=global_addr, [127:126]=type 2.
// D# group1: [17:16]=data_size=1(2B), dim0 @48, dim1 @80, tile0 @112, tile1 @128,
//            dim0_stride @160 (element units). workgroup_mask=0 (not in cluster).
__device__ __forceinline__ void tdm_load_tile_2d(void* lds_dst, const void* gsrc,
                                                 unsigned tensor_dim0, unsigned tensor_dim1,
                                                 unsigned tile_dim0, unsigned tile_dim1,
                                                 unsigned long long stride0) {
  unsigned long long ga = (unsigned long long)gsrc;
  unsigned lds_off = (unsigned)(unsigned long long)lds_dst;
  v4u g0;
  g0[0] = 1u;                                                    // count=1, user D#
  g0[1] = lds_off;                                               // lds_addr
  g0[2] = (unsigned)ga;                                          // global_addr[31:0]
  g0[3] = (unsigned)((ga >> 32) & 0x01FFFFFFull) | (2u << 30);   // addr[56:32] | type=2
  v8u g1;
  g1[0] = (1u << 16);                                            // data_size=2B, mask=0
  g1[1] = (tensor_dim0 & 0xFFFFu) << 16;                         // dim0[15:0] @ bit48
  g1[2] = (tensor_dim0 >> 16) | ((tensor_dim1 & 0xFFFFu) << 16); // dim0[31:16] | dim1[15:0]
  g1[3] = (tensor_dim1 >> 16) | (tile_dim0 << 16);               // dim1[31:16] | tile0
  g1[4] = tile_dim1;                                             // tile1 | tile2=0
  g1[5] = (unsigned)(stride0 & 0xFFFFFFFFull);                   // dim0_stride[31:0]
  g1[6] = (unsigned)((stride0 >> 32) & 0xFFFFull);               // stride0[47:32] | stride1=0
  g1[7] = 0u;
  asm volatile("tensor_load_to_lds %0, %1" :: "s"(g0), "s"(g1) : "memory");
}
__device__ __forceinline__ void wait_tensor0() {
  __builtin_amdgcn_s_wait_tensorcnt(0);
}

// ---------------- Kernel 1: x fp32 -> bf16 ----------------
__global__ void cvt_x(const float* __restrict__ x, bf16* __restrict__ xb, int n) {
  int i = blockIdx.x * blockDim.x + threadIdx.x;
  if (i < n) xb[i] = (bf16)x[i];
}

// ---------------- Kernel 2: fold LoRA into weights, cvt to bf16 ----------------
__global__ void build_weights(const float* __restrict__ Wq, const float* __restrict__ Wk,
                              const float* __restrict__ Wv,
                              const float* __restrict__ qA, const float* __restrict__ qB,
                              const float* __restrict__ vA, const float* __restrict__ vB,
                              bf16* __restrict__ wq, bf16* __restrict__ wk,
                              bf16* __restrict__ wv) {
  int idx = blockIdx.x * blockDim.x + threadIdx.x;  // [0, H*H)
  int o = idx >> 10;
  int i = idx & (H_ - 1);
  float accq = Wq[idx];
  float accv = Wv[idx];
#pragma unroll
  for (int r = 0; r < R_; ++r) {
    accq += LSCALE * qB[o * R_ + r] * qA[r * H_ + i];
    accv += LSCALE * vB[o * R_ + r] * vA[r * H_ + i];
  }
  wq[idx] = (bf16)accq;
  wk[idx] = (bf16)Wk[idx];
  wv[idx] = (bf16)accv;
}

// ---------------- Kernel 3: y = x @ W^T + b, output bf16 in [B,NH,T,DH] ----------------
// Block tile: 128 (M) x 64 (N), 4 waves; TDM stages tiles, 8 WMMAs / wave / K-step.
__global__ __launch_bounds__(128) void proj_gemm(const bf16* __restrict__ xb,
                                                 const bf16* __restrict__ w,
                                                 const float* __restrict__ bias,
                                                 bf16* __restrict__ out) {
  __shared__ alignas(16) bf16 xs[128 * 32];
  __shared__ alignas(16) bf16 wsm[64 * 32];
  const int m0 = blockIdx.x * 128;
  const int n0 = blockIdx.y * 64;
  const int tid = threadIdx.x;
  const int wave = tid >> 5;
  const int lane = tid & 31;
  const int half = lane >> 4;
  const int r = lane & 15;

  v8f acc[2][4];
#pragma unroll
  for (int s = 0; s < 2; ++s)
#pragma unroll
    for (int n = 0; n < 4; ++n)
#pragma unroll
      for (int e = 0; e < 8; ++e) acc[s][n][e] = 0.0f;

  for (int ko = 0; ko < H_; ko += 32) {
    // TDM-stage both tiles: one DMA descriptor each, issued by wave 0.
    if (wave == 0) {
      tdm_load_tile_2d(xs, xb + (size_t)m0 * H_ + ko,
                       /*dim0=*/H_, /*dim1=*/B_ * T_ - m0,
                       /*tile0=*/32, /*tile1=*/128, /*stride0=*/H_);
      tdm_load_tile_2d(wsm, w + (size_t)n0 * H_ + ko,
                       /*dim0=*/H_, /*dim1=*/H_ - n0,
                       /*tile0=*/32, /*tile1=*/64, /*stride0=*/H_);
    }
    // Prefetch next K chunk into GL2 while the TDM runs.
    if (ko + 32 < H_) {
      __builtin_prefetch(xb + (size_t)(m0 + tid) * H_ + ko + 32, 0, 3);
      if (tid < 64) __builtin_prefetch(w + (size_t)(n0 + tid) * H_ + ko + 32, 0, 3);
    }
    if (wave == 0) wait_tensor0();
    __syncthreads();

    // Two A fragments: this wave's 32 rows of x.
    v16bf a[2];
#pragma unroll
    for (int sub = 0; sub < 2; ++sub) {
      int arow = wave * 32 + sub * 16 + r;
#pragma unroll
      for (int v = 0; v < 8; ++v) {
        int kk = kpatA(v, half);
        a[sub][2 * v]     = xs[arow * 32 + kk];
        a[sub][2 * v + 1] = xs[arow * 32 + kk + 1];
      }
    }
#pragma unroll
    for (int n = 0; n < 4; ++n) {
      v16bf b;
      int bcol = n * 16 + r;  // W row n0+bcol is output column
#pragma unroll
      for (int v = 0; v < 8; ++v) {
        int kk = kpatB(v, half);
        b[2 * v]     = wsm[bcol * 32 + kk];
        b[2 * v + 1] = wsm[bcol * 32 + kk + 1];
      }
#pragma unroll
      for (int sub = 0; sub < 2; ++sub)
        acc[sub][n] = __builtin_amdgcn_wmma_f32_16x16x32_bf16(false, a[sub], false, b,
                                                              (short)0, acc[sub][n],
                                                              false, false);
    }
    __syncthreads();
  }

  // Epilogue: bias add + store bf16 in [B, NH, T, DH].
  int mbase = m0 + wave * 32;
#pragma unroll
  for (int n = 0; n < 4; ++n) {
    int o = n0 + n * 16 + r;
    float bi = bias[o];
    int head = o >> 6;
    int d = o & 63;
#pragma unroll
    for (int sub = 0; sub < 2; ++sub) {
#pragma unroll
      for (int g = 0; g < 8; ++g) {
        int mrow = mbase + sub * 16 + g + 8 * half;  // C layout rows
        int bidx = mrow >> 11;                       // / T_
        int t = mrow & (T_ - 1);
        size_t oi = (((size_t)bidx * NH_ + head) * T_ + t) * DH_ + d;
        out[oi] = (bf16)(acc[sub][n][g] + bi);
      }
    }
  }
}

// ---------------- Kernel 4: causal flash attention ----------------
__global__ __launch_bounds__(128) void flash_attn(const bf16* __restrict__ qb,
                                                  const bf16* __restrict__ kb,
                                                  const bf16* __restrict__ vb,
                                                  const float* __restrict__ amask,
                                                  float* __restrict__ out) {
  __shared__ alignas(16) bf16 ksm[32 * 64];     // K chunk, row-major [s][d]
  __shared__ alignas(16) bf16 vtm[64 * 32];     // V chunk, transposed [d][s]
  __shared__ alignas(16) bf16 ps[4 * 16 * 32];  // per-wave P staging
  const int b = blockIdx.z;
  const int h = blockIdx.y;
  const int m0 = blockIdx.x * 64;
  const int tid = threadIdx.x;
  const int wave = tid >> 5;
  const int lane = tid & 31;
  const int half = lane >> 4;
  const int r = lane & 15;

  const bf16* qhead = qb + ((size_t)b * NH_ + h) * T_ * DH_;
  const bf16* khead = kb + ((size_t)b * NH_ + h) * T_ * DH_;
  const bf16* vhead = vb + ((size_t)b * NH_ + h) * T_ * DH_;

  // Q A-fragments (16 rows x 64 d) for this wave: 2 K-steps of 32.
  v16bf aq[2];
  {
    int trow = m0 + wave * 16 + r;
    const bf16* qrow = qhead + (size_t)trow * DH_;
#pragma unroll
    for (int kc = 0; kc < 2; ++kc)
#pragma unroll
      for (int v = 0; v < 8; ++v) {
        int kk = kc * 32 + kpatA(v, half);
        aq[kc][2 * v]     = qrow[kk];
        aq[kc][2 * v + 1] = qrow[kk + 1];
      }
  }

  float mrow[8], lrow[8];
  v8f oacc[4];
#pragma unroll
  for (int g = 0; g < 8; ++g) { mrow[g] = -3.0e38f; lrow[g] = 0.0f; }
#pragma unroll
  for (int n = 0; n < 4; ++n)
#pragma unroll
    for (int e = 0; e < 8; ++e) oacc[n][e] = 0.0f;

  const int send = m0 + 64;  // causal: no keys beyond last query of this tile
  for (int s0 = 0; s0 < send; s0 += 32) {
    // Async-stage K chunk (32x64 bf16 = 256 b128 slots, 2 per thread).
#pragma unroll
    for (int it = 0; it < 2; ++it) {
      int li = tid + it * 128;
      int row = li >> 3;  // 8 uint4 per 64-wide row
      int q = li & 7;
      async_copy_b128(((uint4*)ksm) + li,
                      (const uint4*)(khead + (size_t)(s0 + row) * DH_) + q);
    }
    // Stage V transposed [d][s] (needs VALU transpose path).
#pragma unroll
    for (int it = 0; it < 16; ++it) {
      int e = tid * 16 + it;  // 0..2047
      int s = e >> 6;
      int d = e & 63;
      vtm[d * 32 + s] = vhead[(size_t)(s0 + s) * DH_ + d];
    }
    // Prefetch next chunk's K/V rows.
    if (s0 + 32 < send) {
      __builtin_prefetch(khead + (size_t)(s0 + 32 + (tid >> 2)) * DH_, 0, 3);
      __builtin_prefetch(vhead + (size_t)(s0 + 32 + (tid >> 2)) * DH_, 0, 3);
    }
    wait_async0();
    __syncthreads();

    // Scores: two 16x16 tiles over this 32-key chunk, K-dim = DH (2 wmma steps).
    v8f sacc[2];
#pragma unroll
    for (int st = 0; st < 2; ++st)
#pragma unroll
      for (int e = 0; e < 8; ++e) sacc[st][e] = 0.0f;
#pragma unroll
    for (int st = 0; st < 2; ++st) {
#pragma unroll
      for (int kc = 0; kc < 2; ++kc) {
        v16bf bk;
        int scol = st * 16 + r;
#pragma unroll
        for (int v = 0; v < 8; ++v) {
          int kk = kc * 32 + kpatB(v, half);
          bk[2 * v]     = ksm[scol * 64 + kk];
          bk[2 * v + 1] = ksm[scol * 64 + kk + 1];
        }
        sacc[st] = __builtin_amdgcn_wmma_f32_16x16x32_bf16(false, aq[kc], false, bk,
                                                           (short)0, sacc[st], false, false);
      }
    }

    // Online softmax (rows live in 16-lane groups; shfl_xor 1/2/4/8 reduces a group).
    float p0[8], p1[8];
#pragma unroll
    for (int g = 0; g < 8; ++g) {
      int trow = m0 + wave * 16 + g + 8 * half;
      int sc0 = s0 + r;
      int sc1 = s0 + 16 + r;
      float v0 = sacc[0][g] * 0.125f + amask[(size_t)b * T_ + sc0];
      float v1 = sacc[1][g] * 0.125f + amask[(size_t)b * T_ + sc1];
      if (sc0 > trow) v0 = -1.0e30f;
      if (sc1 > trow) v1 = -1.0e30f;
      float tm = fmaxf(v0, v1);
      tm = fmaxf(tm, __shfl_xor(tm, 1));
      tm = fmaxf(tm, __shfl_xor(tm, 2));
      tm = fmaxf(tm, __shfl_xor(tm, 4));
      tm = fmaxf(tm, __shfl_xor(tm, 8));
      float mnew = fmaxf(mrow[g], tm);
      float scale = __expf(mrow[g] - mnew);
      float e0 = __expf(v0 - mnew);
      float e1 = __expf(v1 - mnew);
      float rs = e0 + e1;
      rs += __shfl_xor(rs, 1);
      rs += __shfl_xor(rs, 2);
      rs += __shfl_xor(rs, 4);
      rs += __shfl_xor(rs, 8);
      lrow[g] = lrow[g] * scale + rs;
      mrow[g] = mnew;
#pragma unroll
      for (int n = 0; n < 4; ++n) oacc[n][g] *= scale;
      p0[g] = e0;
      p1[g] = e1;
    }

    // Re-layout P (C layout -> A fragment) through per-wave LDS staging.
    bf16* pw = ps + wave * 16 * 32;
#pragma unroll
    for (int g = 0; g < 8; ++g) {
      int row = g + 8 * half;
      pw[row * 32 + r]      = (bf16)p0[g];
      pw[row * 32 + 16 + r] = (bf16)p1[g];
    }
    asm volatile("s_wait_dscnt 0x0" ::: "memory");  // same-wave LDS RAW

    v16bf ap;
#pragma unroll
    for (int v = 0; v < 8; ++v) {
      int kk = kpatA(v, half);
      ap[2 * v]     = pw[r * 32 + kk];
      ap[2 * v + 1] = pw[r * 32 + kk + 1];
    }
#pragma unroll
    for (int n = 0; n < 4; ++n) {
      v16bf bvf;
      int dcol = n * 16 + r;
#pragma unroll
      for (int v = 0; v < 8; ++v) {
        int kk = kpatB(v, half);
        bvf[2 * v]     = vtm[dcol * 32 + kk];
        bvf[2 * v + 1] = vtm[dcol * 32 + kk + 1];
      }
      oacc[n] = __builtin_amdgcn_wmma_f32_16x16x32_bf16(false, ap, false, bvf, (short)0,
                                                        oacc[n], false, false);
    }
    __syncthreads();
  }

  // Normalize and store fp32 output [B, T, H] with head interleave.
#pragma unroll
  for (int g = 0; g < 8; ++g) {
    int trow = m0 + wave * 16 + g + 8 * half;
    float inv = 1.0f / lrow[g];
#pragma unroll
    for (int n = 0; n < 4; ++n) {
      int d = n * 16 + r;
      out[((size_t)b * T_ + trow) * H_ + h * DH_ + d] = oacc[n][g] * inv;
    }
  }
}

extern "C" void kernel_launch(void* const* d_in, const int* in_sizes, int n_in,
                              void* d_out, int out_size, void* d_ws, size_t ws_size,
                              hipStream_t stream) {
  const float* x     = (const float*)d_in[0];
  const float* amask = (const float*)d_in[1];
  const float* Wq    = (const float*)d_in[2];
  const float* bq    = (const float*)d_in[3];
  const float* Wk    = (const float*)d_in[4];
  const float* bk    = (const float*)d_in[5];
  const float* Wv    = (const float*)d_in[6];
  const float* bv    = (const float*)d_in[7];
  const float* qA    = (const float*)d_in[8];
  const float* qB    = (const float*)d_in[9];
  const float* vA    = (const float*)d_in[10];
  const float* vB    = (const float*)d_in[11];

  char* ws = (char*)d_ws;
  bf16* xb   = (bf16*)ws; ws += (size_t)B_ * T_ * H_ * 2;
  bf16* wq   = (bf16*)ws; ws += (size_t)H_ * H_ * 2;
  bf16* wk   = (bf16*)ws; ws += (size_t)H_ * H_ * 2;
  bf16* wv   = (bf16*)ws; ws += (size_t)H_ * H_ * 2;
  bf16* qbuf = (bf16*)ws; ws += (size_t)B_ * T_ * H_ * 2;
  bf16* kbuf = (bf16*)ws; ws += (size_t)B_ * T_ * H_ * 2;
  bf16* vbuf = (bf16*)ws; ws += (size_t)B_ * T_ * H_ * 2;

  int nx = B_ * T_ * H_;
  cvt_x<<<(nx + 255) / 256, 256, 0, stream>>>(x, xb, nx);
  build_weights<<<(H_ * H_) / 256, 256, 0, stream>>>(Wq, Wk, Wv, qA, qB, vA, vB, wq, wk, wv);

  dim3 gp(B_ * T_ / 128, H_ / 64);
  proj_gemm<<<gp, 128, 0, stream>>>(xb, wq, bq, qbuf);
  proj_gemm<<<gp, 128, 0, stream>>>(xb, wk, bk, kbuf);
  proj_gemm<<<gp, 128, 0, stream>>>(xb, wv, bv, vbuf);

  dim3 ga(T_ / 64, NH_, B_);
  flash_attn<<<ga, 128, 0, stream>>>(qbuf, kbuf, vbuf, amask, (float*)d_out);
}
